// InvPntAttn_76020921139685
// MI455X (gfx1250) — compile-verified, matrix operand order checked by reading, not run on hardware
//
#include <hip/hip_runtime.h>
#include <hip/hip_bf16.h>
#include <math.h>

// ---------------- problem constants ----------------
#define NL   512          // L
#define DS_  384
#define DP_  256
#define H_   12
#define A_   16
#define Q_   4
#define V_   8
#define PERH 304          // DP + A + 3V + V
#define SHID_ (H_ * PERH) // 3648
#define WC_  0.23570226039551587f   // sqrt(2/(9*Q))
#define WL_  0.57735026918962584f   // sqrt(1/3)

typedef __bf16 bf16;
typedef __attribute__((ext_vector_type(16))) __bf16 bf16x16;
typedef __attribute__((ext_vector_type(8)))  float  floatx8;

// ---------------------------------------------------------------------------
// Fragment loader, all strides/shapes compile-time.
// Layout (per documented CDNA5 16-bit 16x32 A / 32x16 B):
//   lanes 0-15  : fixedIdx = lane,    K in {k0+0..7,  k0+16..23}
//   lanes 16-31 : fixedIdx = lane-16, K in {k0+8..15, k0+24..31}
// Fast path: contiguous K (sK==1), no guards -> 4x float4 (global_load_b128).
// ---------------------------------------------------------------------------
template<long sF, long sK, int FMAX, int KMAX>
__device__ __forceinline__ bf16x16
load_frag(const float* __restrict__ P, int fixedIdx, int k0, int half)
{
    constexpr bool fGuard = (FMAX % 16) != 0;
    constexpr bool kGuard = (KMAX % 32) != 0;
    bf16x16 f;

    if constexpr (!fGuard && !kGuard && sK == 1 && (sF % 4) == 0) {
        // fixedIdx*sF is 4-float aligned, k0 is a multiple of 32, 8*half is 0/8
        const float* base = P + (long)fixedIdx * sF + k0 + 8 * half;
        float4 x0 = *(const float4*)(base + 0);
        float4 x1 = *(const float4*)(base + 4);
        float4 x2 = *(const float4*)(base + 16);
        float4 x3 = *(const float4*)(base + 20);
        f[0]  = (bf16)x0.x; f[1]  = (bf16)x0.y; f[2]  = (bf16)x0.z; f[3]  = (bf16)x0.w;
        f[4]  = (bf16)x1.x; f[5]  = (bf16)x1.y; f[6]  = (bf16)x1.z; f[7]  = (bf16)x1.w;
        f[8]  = (bf16)x2.x; f[9]  = (bf16)x2.y; f[10] = (bf16)x2.z; f[11] = (bf16)x2.w;
        f[12] = (bf16)x3.x; f[13] = (bf16)x3.y; f[14] = (bf16)x3.z; f[15] = (bf16)x3.w;
        return f;
    }

    bool rv = true;
    if constexpr (fGuard) rv = (fixedIdx < FMAX);
    const long base = (long)fixedIdx * sF;
#pragma unroll
    for (int i = 0; i < 8; ++i) {
        const int ka = k0 + 8 * half + i;
        const int kb = ka + 16;
        bool va = rv, vb = rv;
        if constexpr (kGuard) { va = va && (ka < KMAX); vb = vb && (kb < KMAX); }
        long ia = base + (long)ka * sK;
        long ib = base + (long)kb * sK;
        if constexpr (fGuard || kGuard) { ia = va ? ia : 0; ib = vb ? ib : 0; }
        float xa = P[ia];
        float xb = P[ib];
        if constexpr (fGuard || kGuard) { xa = va ? xa : 0.0f; xb = vb ? xb : 0.0f; }
        f[i]     = (bf16)xa;
        f[i + 8] = (bf16)xb;
    }
    return f;
}

// ---------------------------------------------------------------------------
// Generic strided batched GEMM, everything compile-time except pointers/alpha:
//   C[b](m,n) = alpha * sum_k A[b](m,k) * B[b](k,n)   (+bias[n]) (+addC) (relu)
// A(row,k) = A[b*sAb + row*sAr + k*sAk]
// B(k,col) = B[b*sBb + k*sBk  + col*sBn]
// C(m,n)   = C[b*sCb + m*sCr  + n*sCn]     addC indexed like C with sDb batch
// One wave32 computes a TM x TN block of 16x16 tiles: TM*TN WMMAs per K-step,
// each A fragment reused TN times, each B fragment TM times.
// blockDim = 128 (4 waves); tile id is wave-uniform -> EXEC all ones at WMMA.
// ---------------------------------------------------------------------------
template<long sAr, long sAk, long sAb,
         long sBk, long sBn, long sBb,
         long sCr, long sCn, long sCb, long sDb,
         int M, int N, int K, int TM, int TN,
         bool BIAS, bool ADD, bool RELU>
__global__ __launch_bounds__(128) void gemm_bf16_wmma(
    const float* __restrict__ A, const float* __restrict__ B,
    float* __restrict__ C, const float* __restrict__ bias,
    const float* __restrict__ addC, float alpha)
{
    static_assert(TM == 1 || (M % (16 * TM)) == 0, "TM must tile M evenly");
    static_assert(TN == 1 || (N % (16 * TN)) == 0, "TN must tile N evenly");
    constexpr int ntn   = (N + 16 * TN - 1) / (16 * TN);
    constexpr int ntm   = (M + 16 * TM - 1) / (16 * TM);
    constexpr int tiles = ntm * ntn;

    const int tile = blockIdx.x * 4 + (threadIdx.x >> 5);
    if constexpr ((tiles % 4) != 0) {
        if (tile >= tiles) return;              // wave-uniform exit
    }
    const int tm   = tile / ntn;
    const int tn   = tile % ntn;
    const int lane = threadIdx.x & 31;
    const int lo   = lane & 15;
    const int half = lane >> 4;

    const int batch = blockIdx.y;
    const float* Ab = A + (long)batch * sAb;
    const float* Bb = B + (long)batch * sBb;

    const int row0 = tm * 16 * TM + lo;
    const int col0 = tn * 16 * TN + lo;

    floatx8 acc[TM][TN];
#pragma unroll
    for (int im = 0; im < TM; ++im)
#pragma unroll
        for (int in = 0; in < TN; ++in)
            acc[im][in] = floatx8{};

    for (int k0 = 0; k0 < K; k0 += 32) {
        if constexpr (K > 32) {                 // CDNA5 speculative prefetch
            if (k0 + 32 < K) {
                __builtin_prefetch(&Ab[(long)row0 * sAr + (long)(k0 + 32) * sAk], 0, 0);
                __builtin_prefetch(&Bb[(long)col0 * sBn + (long)(k0 + 32) * sBk], 0, 0);
            }
        }
        bf16x16 fa[TM], fb[TN];
#pragma unroll
        for (int im = 0; im < TM; ++im)
            fa[im] = load_frag<sAr, sAk, M, K>(Ab, row0 + im * 16, k0, half);
#pragma unroll
        for (int in = 0; in < TN; ++in)
            fb[in] = load_frag<sBn, sBk, N, K>(Bb, col0 + in * 16, k0, half);
#pragma unroll
        for (int im = 0; im < TM; ++im)
#pragma unroll
            for (int in = 0; in < TN; ++in)
                acc[im][in] = __builtin_amdgcn_wmma_f32_16x16x32_bf16(
                    false, fa[im], false, fb[in], (short)0, acc[im][in], false, false);
    }

    float*       Cb = C + (long)batch * sCb;
    const float* Db = addC + (long)batch * sDb;
#pragma unroll
    for (int im = 0; im < TM; ++im) {
#pragma unroll
        for (int in = 0; in < TN; ++in) {
#pragma unroll
            for (int r = 0; r < 8; ++r) {
                const int m = tm * 16 * TM + im * 16 + r + 8 * half;
                const int n = tn * 16 * TN + in * 16 + lo;
                bool ok = true;
                if constexpr ((M % 16) != 0) ok = ok && (m < M);
                if constexpr ((N % 16) != 0) ok = ok && (n < N);
                if (ok) {
                    float v = alpha * acc[im][in][r];
                    if constexpr (BIAS) v += bias[n];
                    if constexpr (ADD)  v += Db[(long)m * sCr + (long)n * sCn];
                    if constexpr (RELU) v = fmaxf(v, 0.0f);
                    Cb[(long)m * sCr + (long)n * sCn] = v;
                }
            }
        }
    }
}

template<long sAr, long sAk, long sAb,
         long sBk, long sBn, long sBb,
         long sCr, long sCn, long sCb, long sDb,
         int M, int N, int K, int TM, int TN,
         bool BIAS, bool ADD, bool RELU>
static inline void launch_gemm(hipStream_t st,
                               const float* A, const float* B, float* C,
                               const float* bias, const float* addC,
                               int batches, float alpha = 1.0f)
{
    constexpr int tiles = ((M + 16 * TM - 1) / (16 * TM)) * ((N + 16 * TN - 1) / (16 * TN));
    dim3 grid((tiles + 3) / 4, batches);
    gemm_bf16_wmma<sAr, sAk, sAb, sBk, sBn, sBb, sCr, sCn, sCb, sDb,
                   M, N, K, TM, TN, BIAS, ADD, RELU>
        <<<grid, dim3(128), 0, st>>>(A, B, C, bias, addC, alpha);
}

// ---------------------------------------------------------------------------
// Quaternion -> rotation; apply rot+t in place to qp/kp/vp local points.
// ---------------------------------------------------------------------------
__device__ __forceinline__ void quat_rot(const float* quat, int l, float R[9])
{
    float qx = quat[l * 3 + 0], qy = quat[l * 3 + 1], qz = quat[l * 3 + 2];
    float inv = rsqrtf(1.0f + qx * qx + qy * qy + qz * qz);
    float w = inv, x = qx * inv, y = qy * inv, z = qz * inv;
    R[0] = 1.f - 2.f * (y * y + z * z); R[1] = 2.f * (x * y - w * z); R[2] = 2.f * (x * z + w * y);
    R[3] = 2.f * (x * y + w * z); R[4] = 1.f - 2.f * (x * x + z * z); R[5] = 2.f * (y * z - w * x);
    R[6] = 2.f * (x * z - w * y); R[7] = 2.f * (y * z + w * x); R[8] = 1.f - 2.f * (x * x + y * y);
}

__global__ void rot_points_kernel(const float* __restrict__ quat,
                                  const float* __restrict__ trsl,
                                  float* __restrict__ qp,
                                  float* __restrict__ kp,
                                  float* __restrict__ vp)
{
    int l = blockIdx.x;
    int t = threadIdx.x;                // 0..191
    float R[9];
    quat_rot(quat, l, R);
    float tx = trsl[l * 3 + 0], ty = trsl[l * 3 + 1], tz = trsl[l * 3 + 2];

    float* base;
    int idx;
    if (t < 48)       { base = qp + (long)l * (H_ * Q_ * 3); idx = t; }
    else if (t < 96)  { base = kp + (long)l * (H_ * Q_ * 3); idx = t - 48; }
    else              { base = vp + (long)l * (H_ * V_ * 3); idx = t - 96; }

    float px = base[idx * 3 + 0], py = base[idx * 3 + 1], pz = base[idx * 3 + 2];
    base[idx * 3 + 0] = R[0] * px + R[1] * py + R[2] * pz + tx;
    base[idx * 3 + 1] = R[3] * px + R[4] * py + R[5] * pz + ty;
    base[idx * 3 + 2] = R[6] * px + R[7] * py + R[8] * pz + tz;
}

// ---------------------------------------------------------------------------
// Fused scores + softmax.  grid = (L, H), block = 256 (2 j's per thread).
// logits = WL*(qk/4 + b - coef_h * ||qp_g[i]-kp_g[j]||^2); softmax over j.
// Written in place over the qk buffer (becomes `a`).
// ---------------------------------------------------------------------------
__global__ void scores_softmax_kernel(float* __restrict__ qk,
                                      const float* __restrict__ bbuf,
                                      const float* __restrict__ qp_g,
                                      const float* __restrict__ kp_g,
                                      const float* __restrict__ scale)
{
    const int i = blockIdx.x, h = blockIdx.y, tid = threadIdx.x;
    __shared__ float qpv[12];
    __shared__ float red[256];
    __shared__ float coef_s;

    if (tid < 12) qpv[tid] = qp_g[(long)i * (H_ * Q_ * 3) + h * 12 + tid];
    if (tid == 0) {
        float sc = scale[h];
        float sp = (sc > 20.0f) ? sc : log1pf(expf(sc));
        coef_s = WC_ * sp * 0.5f;
    }
    __syncthreads();
    const float coef = coef_s;

    float lg[2];
#pragma unroll
    for (int it = 0; it < 2; ++it) {
        int j = tid + it * 256;
        const float* kpr = kp_g + (long)j * (H_ * Q_ * 3) + h * 12;
        float d2 = 0.0f;
#pragma unroll
        for (int c = 0; c < 12; ++c) { float d = qpv[c] - kpr[c]; d2 += d * d; }
        long off = ((long)i * NL + j) * H_ + h;
        lg[it] = WL_ * (qk[off] + bbuf[off] - coef * d2);
    }

    float m = fmaxf(lg[0], lg[1]);
    red[tid] = m; __syncthreads();
    for (int s = 128; s > 0; s >>= 1) { if (tid < s) red[tid] = fmaxf(red[tid], red[tid + s]); __syncthreads(); }
    m = red[0]; __syncthreads();

    float e0 = __expf(lg[0] - m), e1 = __expf(lg[1] - m);
    red[tid] = e0 + e1; __syncthreads();
    for (int s = 128; s > 0; s >>= 1) { if (tid < s) red[tid] += red[tid + s]; __syncthreads(); }
    float inv = 1.0f / red[0];

    qk[((long)i * NL + tid      ) * H_ + h] = e0 * inv;
    qk[((long)i * NL + tid + 256) * H_ + h] = e1 * inv;
}

// ---------------------------------------------------------------------------
// Finalize shid: ovp = R^T (ovp_g - t), plus per-point norms.
// ---------------------------------------------------------------------------
__global__ void finalize_kernel(const float* __restrict__ quat,
                                const float* __restrict__ trsl,
                                float* __restrict__ shid)
{
    int gid = blockIdx.x * blockDim.x + threadIdx.x;
    if (gid >= NL * H_ * V_) return;
    int v = gid % V_;
    int h = (gid / V_) % H_;
    int l = gid / (V_ * H_);

    float R[9];
    quat_rot(quat, l, R);
    float tx = trsl[l * 3 + 0], ty = trsl[l * 3 + 1], tz = trsl[l * 3 + 2];

    float* p = shid + (long)l * SHID_ + h * PERH + (DP_ + A_) + v * 3;
    float gx = p[0] - tx, gy = p[1] - ty, gz = p[2] - tz;
    float ox = R[0] * gx + R[3] * gy + R[6] * gz;   // R^T
    float oy = R[1] * gx + R[4] * gy + R[7] * gz;
    float oz = R[2] * gx + R[5] * gy + R[8] * gz;
    p[0] = ox; p[1] = oy; p[2] = oz;
    shid[(long)l * SHID_ + h * PERH + (DP_ + A_ + 3 * V_) + v] =
        sqrtf(ox * ox + oy * oy + oz * oz);
}

// ---------------------------------------------------------------------------
// LayerNorm over DS=384.  grid = L, block = 128 (3 elements/thread).
// ---------------------------------------------------------------------------
__global__ void layernorm_kernel(const float* __restrict__ x,
                                 const float* __restrict__ g,
                                 const float* __restrict__ be,
                                 float* __restrict__ y)
{
    const int l = blockIdx.x, tid = threadIdx.x;
    __shared__ float red[128];
    float v0 = x[(long)l * DS_ + tid];
    float v1 = x[(long)l * DS_ + tid + 128];
    float v2 = x[(long)l * DS_ + tid + 256];

    red[tid] = v0 + v1 + v2; __syncthreads();
    for (int s = 64; s > 0; s >>= 1) { if (tid < s) red[tid] += red[tid + s]; __syncthreads(); }
    float mean = red[0] * (1.0f / DS_); __syncthreads();

    float d0 = v0 - mean, d1 = v1 - mean, d2 = v2 - mean;
    red[tid] = d0 * d0 + d1 * d1 + d2 * d2; __syncthreads();
    for (int s = 64; s > 0; s >>= 1) { if (tid < s) red[tid] += red[tid + s]; __syncthreads(); }
    float inv = rsqrtf(red[0] * (1.0f / DS_) + 1e-5f);

    y[(long)l * DS_ + tid]       = d0 * inv * g[tid]       + be[tid];
    y[(long)l * DS_ + tid + 128] = d1 * inv * g[tid + 128] + be[tid + 128];
    y[(long)l * DS_ + tid + 256] = d2 * inv * g[tid + 256] + be[tid + 256];
}

// ---------------------------------------------------------------------------
extern "C" void kernel_launch(void* const* d_in, const int* in_sizes, int n_in,
                              void* d_out, int out_size, void* d_ws, size_t ws_size,
                              hipStream_t stream)
{
    (void)in_sizes; (void)n_in; (void)out_size; (void)ws_size;
    const float* sfea  = (const float*)d_in[0];
    const float* pfea  = (const float*)d_in[1];
    const float* quat  = (const float*)d_in[2];
    const float* trsl  = (const float*)d_in[3];
    const float* Wq    = (const float*)d_in[4];
    const float* Wk    = (const float*)d_in[5];
    const float* Wv    = (const float*)d_in[6];
    const float* Wqp   = (const float*)d_in[7];
    const float* Wkp   = (const float*)d_in[8];
    const float* Wvp   = (const float*)d_in[9];
    const float* Wb    = (const float*)d_in[10];
    const float* scale = (const float*)d_in[11];
    const float* Ws    = (const float*)d_in[12];
    const float* bs    = (const float*)d_in[13];
    const float* W1    = (const float*)d_in[14];
    const float* b1    = (const float*)d_in[15];
    const float* W2    = (const float*)d_in[16];
    const float* b2    = (const float*)d_in[17];
    const float* W3    = (const float*)d_in[18];
    const float* b3    = (const float*)d_in[19];
    const float* g1    = (const float*)d_in[20];
    const float* be1   = (const float*)d_in[21];
    const float* g2    = (const float*)d_in[22];
    const float* be2   = (const float*)d_in[23];
    float* out = (float*)d_out;

    // workspace layout (floats)
    float* ws = (float*)d_ws;
    float* q    = ws;                 // 512*192
    float* k    = q    + 98304;
    float* v    = k    + 98304;
    float* qp   = v    + 98304;       // 512*144  (rotated in place)
    float* kp   = qp   + 73728;
    float* vp   = kp   + 73728;       // 512*288
    float* qka  = vp   + 147456;      // 512*512*12  qk scores, then `a`
    float* bb   = qka  + 3145728;     // 512*512*12  pair bias b
    float* shid = bb   + 3145728;     // 512*3648
    float* s0   = shid + 1867776;     // 512*384
    float* s1   = s0   + 196608;
    float* h1   = s1   + 196608;
    float* h2   = h1   + 196608;
    float* s2   = h2   + 196608;

    // 1) projections: X(512x384) @ W(384xN)
    launch_gemm<DS_,1,0, 192,1,0, 192,1,0, 0, NL,192,DS_, 2,2, false,false,false>(stream, sfea, Wq,  q,  nullptr, nullptr, 1);
    launch_gemm<DS_,1,0, 192,1,0, 192,1,0, 0, NL,192,DS_, 2,2, false,false,false>(stream, sfea, Wk,  k,  nullptr, nullptr, 1);
    launch_gemm<DS_,1,0, 192,1,0, 192,1,0, 0, NL,192,DS_, 2,2, false,false,false>(stream, sfea, Wv,  v,  nullptr, nullptr, 1);
    launch_gemm<DS_,1,0, 144,1,0, 144,1,0, 0, NL,144,DS_, 2,1, false,false,false>(stream, sfea, Wqp, qp, nullptr, nullptr, 1);
    launch_gemm<DS_,1,0, 144,1,0, 144,1,0, 0, NL,144,DS_, 2,1, false,false,false>(stream, sfea, Wkp, kp, nullptr, nullptr, 1);
    launch_gemm<DS_,1,0, 288,1,0, 288,1,0, 0, NL,288,DS_, 2,2, false,false,false>(stream, sfea, Wvp, vp, nullptr, nullptr, 1);

    // 2) rotate local points to global frame (in place)
    rot_points_kernel<<<dim3(NL), dim3(192), 0, stream>>>(quat, trsl, qp, kp, vp);

    // 3) pair bias: b = pfea(262144x256) @ Wb(256x12)  — pfea streamed once
    launch_gemm<DP_,1,0, H_,1,0, H_,1,0, 0, NL*NL,H_,DP_, 2,1, false,false,false>(stream, pfea, Wb, bb, nullptr, nullptr, 1);

    // 4) qk[i,j,h] = (q_h @ k_h^T)/4 : batch over h
    launch_gemm<192,1,16,              // A(i,d) = q[i*192 + h*16 + d]
                1,192,16,              // B(d,j) = k[j*192 + h*16 + d]  (transposed)
                6144,12,1, 0,          // C(i,j) -> qka[(i*512+j)*12 + h]
                NL,NL,16, 2,2, false,false,false>(stream, q, k, qka, nullptr, nullptr, H_, 0.25f);

    // 5) fused scores + softmax  (qka becomes attention `a`)
    scores_softmax_kernel<<<dim3(NL, H_), dim3(256), 0, stream>>>(qka, bb, qp, kp, scale);

    // 6) op[i,h,p] = sum_j a[i,j,h] pfea[i,j,p] : batch over i — pfea streamed once
    launch_gemm<1,12,6144,             // A(h,j) = a[i*6144 + j*12 + h]
                256,1,131072,          // B(j,p) = pfea[i*131072 + j*256 + p]
                304,1,3648, 0,         // C(h,p) -> shid[i*3648 + h*304 + p]
                H_,DP_,NL, 1,2, false,false,false>(stream, qka, pfea, shid, nullptr, nullptr, NL);

    // 7) ov[i,h,d] = sum_j a[i,j,h] v[j,h,d] : batch over h
    launch_gemm<6144,12,1,             // A(i,j) = a[(i*512+j)*12 + h]
                192,1,16,              // B(j,d) = v[j*192 + h*16 + d]
                3648,1,304, 0,         // C(i,d) -> shid[i*3648 + h*304 + 256 + d]
                NL,A_,NL, 2,1, false,false,false>(stream, qka, v, shid + DP_, nullptr, nullptr, H_);

    // 8) ovp_g[i,h,e] = sum_j a[i,j,h] vp_g[j,h,e] : batch over h, N=24
    launch_gemm<6144,12,1,
                288,1,24,
                3648,1,304, 0,
                NL,V_*3,NL, 1,1, false,false,false>(stream, qka, vp, shid + DP_ + A_, nullptr, nullptr, H_);

    // 9) inverse-rotate ovp, compute norms
    finalize_kernel<<<dim3((NL*H_*V_ + 255) / 256), dim3(256), 0, stream>>>(quat, trsl, shid);

    // 10) s0 = sfea + shid @ Ws + bs
    launch_gemm<SHID_,1,0, DS_,1,0, DS_,1,0, 0, NL,DS_,SHID_, 2,2, true,true,false>(stream, shid, Ws, s0, bs, sfea, 1);

    // 11) LN1
    layernorm_kernel<<<dim3(NL), dim3(128), 0, stream>>>(s0, g1, be1, s1);

    // 12) MLP
    launch_gemm<DS_,1,0, DS_,1,0, DS_,1,0, 0, NL,DS_,DS_, 2,2, true,false,true>(stream, s1, W1, h1, b1, nullptr, 1);
    launch_gemm<DS_,1,0, DS_,1,0, DS_,1,0, 0, NL,DS_,DS_, 2,2, true,false,true>(stream, h1, W2, h2, b2, nullptr, 1);
    launch_gemm<DS_,1,0, DS_,1,0, DS_,1,0, 0, NL,DS_,DS_, 2,2, true,true,false>(stream, h2, W3, s2, b3, s1, 1);

    // 13) LN2 -> output
    layernorm_kernel<<<dim3(NL), dim3(128), 0, stream>>>(s2, g2, be2, out);
}
